// IsingTransformerLayer_16673063043203
// MI455X (gfx1250) — compile-verified, matrix operand order checked by reading, not run on hardware
//
#include <hip/hip_runtime.h>
#include <stdint.h>

// ---------------- types & helpers ----------------
typedef __attribute__((ext_vector_type(16))) __bf16       v16bf;
typedef __attribute__((ext_vector_type(8)))  float        v8f;
typedef __attribute__((ext_vector_type(4)))  unsigned int v4u;

__device__ __forceinline__ unsigned short f2bf(float f) {
    union { float f; uint32_t u; } c; c.f = f;
    uint32_t u = c.u;
    uint32_t r = (u + 0x7FFFu + ((u >> 16) & 1u)) >> 16;   // RNE
    return (unsigned short)r;
}

union BFrag { v16bf v; v4u q[2]; };

// Load a 16x32 bf16 WMMA operand fragment.
// A operand: row = m-tile base, matrix row-major (lda elems).
// B operand: pass the TRANSPOSED matrix (B^T row-major), row = n-tile base.
// Lane L: r = row + (L&15); kb = k0 + (L<16?0:8);
// elems 0..7 = K kb..kb+7, elems 8..15 = K kb+16..kb+23  (two b128 loads).
__device__ __forceinline__ v16bf load_frag(const unsigned short* __restrict__ base,
                                           int ld, int row, int k0) {
    int lane = threadIdx.x & 31;
    int r  = row + (lane & 15);
    int kb = k0 + ((lane & 16) ? 8 : 0);
    const unsigned short* p = base + (size_t)r * ld + kb;
    BFrag f;
    f.q[0] = *(const v4u*)(p);
    f.q[1] = *(const v4u*)(p + 16);
    return f.v;
}

__device__ __forceinline__ v8f wmma_bf16(v16bf a, v16bf b, v8f c) {
    return __builtin_amdgcn_wmma_f32_16x16x32_bf16(false, a, false, b,
                                                   (short)0, c, false, false);
}

// C/D f32 16x16 layout: lane L: col n = L&15; rows m = v + (L<16?0:8)
__device__ __forceinline__ void store_ctile_f32(float* __restrict__ C, int ldc,
                                                int m0, int n0, v8f acc) {
    int lane = threadIdx.x & 31;
    int n  = n0 + (lane & 15);
    int mb = m0 + ((lane & 16) ? 8 : 0);
#pragma unroll
    for (int v = 0; v < 8; ++v) C[(size_t)(mb + v) * ldc + n] = acc[v];
}

// ---------------- constants ----------------
#define NN   1024
#define DIM  512
#define HEADS 8
#define DH   64

// ---------------- kernels ----------------

// LayerNorm + /sqrt(dim_head); writes f32 master + bf16 copy
__global__ void ln_kernel(const float* __restrict__ x, const float* __restrict__ w,
                          const float* __restrict__ b, float* __restrict__ xnf,
                          unsigned short* __restrict__ xnb) {
    int row = blockIdx.x;
    int tid = threadIdx.x;
    __shared__ float red[2][8];
    const float* xr = x + (size_t)row * DIM;
    float s = 0.f, s2 = 0.f;
    for (int j = tid; j < DIM; j += 256) { float v = xr[j]; s += v; s2 += v * v; }
    for (int off = 16; off >= 1; off >>= 1) { s += __shfl_xor(s, off, 32); s2 += __shfl_xor(s2, off, 32); }
    int wv = tid >> 5;
    if ((tid & 31) == 0) { red[0][wv] = s; red[1][wv] = s2; }
    __syncthreads();
    if (tid == 0) {
        float a = 0.f, a2 = 0.f;
        for (int k = 0; k < 8; ++k) { a += red[0][k]; a2 += red[1][k]; }
        red[0][0] = a; red[1][0] = a2;
    }
    __syncthreads();
    float mean = red[0][0] * (1.f / DIM);
    float var  = red[1][0] * (1.f / DIM) - mean * mean;
    float inv  = rsqrtf(var + 1e-5f);
    for (int j = tid; j < DIM; j += 256) {
        float v = ((xr[j] - mean) * inv * w[j] + b[j]) * 0.125f;  // / sqrt(64)
        xnf[(size_t)row * DIM + j] = v;
        xnb[(size_t)row * DIM + j] = f2bf(v);
    }
}

// Transpose Wq, Wk to bf16 (B^T copies for WMMA B operands)
__global__ void transw_kernel(const float* __restrict__ Wq, const float* __restrict__ Wk,
                              unsigned short* __restrict__ WqT, unsigned short* __restrict__ WkT) {
    int idx = blockIdx.x * 256 + threadIdx.x;      // 2 * 512 * 512
    int sel = idx >> 18;
    int r = (idx >> 9) & 511;
    int c = idx & 511;
    const float* W = sel ? Wk : Wq;
    unsigned short* WT = sel ? WkT : WqT;
    WT[(size_t)r * DIM + c] = f2bf(W[(size_t)c * DIM + r]);
}

// q = xn@Wq * 0.125 ; k = xn@Wk  (M=1024, N=512, K=512), one wave per 16x16 tile
__global__ void qk_gemm_kernel(const unsigned short* __restrict__ xnb,
                               const unsigned short* __restrict__ WqT,
                               const unsigned short* __restrict__ WkT,
                               unsigned short* __restrict__ qb,
                               unsigned short* __restrict__ kb) {
    int wid = blockIdx.x * 8 + (threadIdx.x >> 5);
    int sel = wid >> 11;                 // 2048 tiles per output matrix
    int tt  = wid & 2047;
    int mt  = tt >> 5;                   // 64 row tiles
    int nt  = tt & 31;                   // 32 col tiles
    const unsigned short* BT = sel ? WkT : WqT;
    v8f acc = {};
    for (int k0 = 0; k0 < DIM; k0 += 32) {
        v16bf a  = load_frag(xnb, DIM, mt * 16, k0);
        v16bf bb = load_frag(BT,  DIM, nt * 16, k0);
        acc = wmma_bf16(a, bb, acc);
    }
    float scale = sel ? 1.f : 0.125f;    // q *= dim_head^-0.5
    unsigned short* outp = sel ? kb : qb;
    int lane = threadIdx.x & 31;
    int n  = nt * 16 + (lane & 15);
    int mb = mt * 16 + ((lane & 16) ? 8 : 0);
#pragma unroll
    for (int v = 0; v < 8; ++v)
        outp[(size_t)(mb + v) * DIM + n] = f2bf(acc[v] * scale);
}

// Per (head, 16-row strip): S = q_h @ k_h^T -> causal softmax -> J = sm/256
// writes J (row-major) and J^T (bf16) + diag(J) f32
__global__ void attn_kernel(const unsigned short* __restrict__ qb,
                            const unsigned short* __restrict__ kb,
                            unsigned short* __restrict__ Jb,
                            unsigned short* __restrict__ JTb,
                            float* __restrict__ Jdiag) {
    __shared__ float S[16 * NN];         // 64KB strip (fits 320KB WGP LDS)
    int hd    = blockIdx.x >> 6;
    int strip = blockIdx.x & 63;
    int m0 = strip * 16;
    const unsigned short* qh = qb + hd * DH;   // row stride DIM
    const unsigned short* kh = kb + hd * DH;
    int wv = threadIdx.x >> 5;
    for (int nt = wv; nt < 64; nt += 8) {      // 64 col tiles, 8 waves
        v8f acc = {};
        for (int k0 = 0; k0 < DH; k0 += 32) {
            v16bf a  = load_frag(qh, DIM, m0,      k0);
            v16bf bf = load_frag(kh, DIM, nt * 16, k0);   // (k^T)^T = k rows
            acc = wmma_bf16(a, bf, acc);
        }
        int lane = threadIdx.x & 31;
        int n  = nt * 16 + (lane & 15);
        int mb = (lane & 16) ? 8 : 0;
#pragma unroll
        for (int v = 0; v < 8; ++v) S[(mb + v) * NN + n] = acc[v];
    }
    __syncthreads();
    // causal softmax, 16 threads per row
    int r = threadIdx.x >> 4;
    int c = threadIdx.x & 15;
    int i = m0 + r;
    float mx = -1e30f;
    for (int j = c; j <= i; j += 16) mx = fmaxf(mx, S[r * NN + j]);
    for (int off = 8; off >= 1; off >>= 1) mx = fmaxf(mx, __shfl_xor(mx, off, 16));
    float sum = 0.f;
    for (int j = c; j <= i; j += 16) sum += __expf(S[r * NN + j] - mx);
    for (int off = 8; off >= 1; off >>= 1) sum += __shfl_xor(sum, off, 16);
    float inv = 1.f / (sum * 256.f);     // / sqrt(n*dim_head)
    size_t hb = (size_t)hd << 20;
    for (int j = c; j < NN; j += 16) {
        float val = (j <= i) ? __expf(S[r * NN + j] - mx) * inv : 0.f;
        unsigned short bv = f2bf(val);
        Jb [hb + (size_t)i * NN + j] = bv;
        JTb[hb + (size_t)j * NN + i] = bv;
        if (j == i) Jdiag[hd * NN + i] = val;
    }
}

__global__ void init_t_kernel(float* __restrict__ t) {
    int idx = blockIdx.x * 256 + threadIdx.x;
    if (idx < HEADS * NN) t[idx] = 1.f;
}

// g^T[hd][d][n] = h[n][d] / t[n]   (bf16, B^T operand for both J sweeps)
__global__ void prep_g_kernel(const float* __restrict__ xnf, const float* __restrict__ t,
                              unsigned short* __restrict__ gT) {
    int idx = blockIdx.x * 256 + threadIdx.x;   // 8*1024*64
    int d  = idx & 63;
    int i  = (idx >> 6) & 1023;
    int hd = idx >> 16;
    float it = 1.f / t[hd * NN + i];
    float h  = xnf[(size_t)i * DIM + hd * DH + d];
    gT[((size_t)hd << 16) + (size_t)d * NN + i] = f2bf(it * h);
}

// Aout = J@g, Bout = J^T@g per head. 2x4 register-blocked: each wave owns
// 2 m-tiles x full N (4 n-tiles): 12 b128 loads per 8 WMMAs per k-chunk.
__global__ void matvec_kernel(const unsigned short* __restrict__ Jb,
                              const unsigned short* __restrict__ JTb,
                              const unsigned short* __restrict__ gT,
                              float* __restrict__ Aout, float* __restrict__ Bout) {
    int wid = blockIdx.x * 8 + (threadIdx.x >> 5);   // 0..511
    int sel = wid >> 8;                              // J vs J^T
    int tt  = wid & 255;
    int hd  = tt >> 5;                               // 0..7
    int mb  = tt & 31;                               // 32-row block
    const unsigned short* A  = (sel ? JTb : Jb) + ((size_t)hd << 20);
    const unsigned short* BT = gT + ((size_t)hd << 16);
    int m0 = mb * 32;
    v8f acc[2][4] = {};
    for (int k0 = 0; k0 < NN; k0 += 32) {
        if (k0 + 64 < NN)   // keep next J chunk flowing from L2
            __builtin_prefetch(A + (size_t)(m0 + (threadIdx.x & 31)) * NN + k0 + 64, 0, 1);
        v16bf a0 = load_frag(A, NN, m0,      k0);
        v16bf a1 = load_frag(A, NN, m0 + 16, k0);
#pragma unroll
        for (int nt = 0; nt < 4; ++nt) {
            v16bf b = load_frag(BT, NN, nt * 16, k0);
            acc[0][nt] = wmma_bf16(a0, b, acc[0][nt]);
            acc[1][nt] = wmma_bf16(a1, b, acc[1][nt]);
        }
    }
    float* C = (sel ? Bout : Aout) + ((size_t)hd << 16);
#pragma unroll
    for (int mi = 0; mi < 2; ++mi)
#pragma unroll
        for (int nt = 0; nt < 4; ++nt)
            store_ctile_f32(C, DH, m0 + mi * 16, nt * 16, acc[mi][nt]);
}

// f(t) = jac_phi(t) + t, Anderson(m=2) update. One block per head.
__global__ void solve_kernel(const float* __restrict__ xnf, const float* __restrict__ Jdiag,
                             const float* __restrict__ Aout, const float* __restrict__ Bout,
                             float* __restrict__ t,
                             float* __restrict__ X0, float* __restrict__ X1,
                             float* __restrict__ F0, float* __restrict__ F1,
                             int mode) {                 // 0 = init history, 1 = step
    int hd = blockIdx.x;
    __shared__ float red[3][8];
    __shared__ float alphas[2];
    float d00 = 0.f, d01 = 0.f, d11 = 0.f;
    for (int rr = 0; rr < 4; ++rr) {
        int i  = threadIdx.x + rr * 256;
        int gi = hd * NN + i;
        float tv = t[gi];
        float it = 1.f / tv;
        const float* ar = Aout + ((size_t)hd << 16) + (size_t)i * DH;
        const float* br = Bout + ((size_t)hd << 16) + (size_t)i * DH;
        const float* hr = xnf + (size_t)i * DIM + hd * DH;
        float s = 0.f;
#pragma unroll 8
        for (int d = 0; d < DH; ++d) { float h = hr[d]; s += (h + ar[d]) * (h + br[d]); }
        float quad = it * it * s;
        float dvi  = it + it * it * Jdiag[gi];
        float ft   = 1.f - 0.5f * dvi - 0.25f * quad + tv;
        if (mode == 0) {
            X0[gi] = tv; X1[gi] = tv; F0[gi] = ft; F1[gi] = ft;
        } else {
            float x0 = X1[gi]; X0[gi] = x0; X1[gi] = tv;
            float f0 = F1[gi]; F0[gi] = f0; F1[gi] = ft;
            float r0 = f0 - x0, r1 = ft - tv;
            d00 += r0 * r0; d01 += r0 * r1; d11 += r1 * r1;
        }
    }
    if (mode == 0) return;
    for (int off = 16; off >= 1; off >>= 1) {
        d00 += __shfl_xor(d00, off, 32);
        d01 += __shfl_xor(d01, off, 32);
        d11 += __shfl_xor(d11, off, 32);
    }
    int wv = threadIdx.x >> 5;
    if ((threadIdx.x & 31) == 0) { red[0][wv] = d00; red[1][wv] = d01; red[2][wv] = d11; }
    __syncthreads();
    if (threadIdx.x == 0) {
        float g00 = 0.f, g01 = 0.f, g11 = 0.f;
        for (int k = 0; k < 8; ++k) { g00 += red[0][k]; g01 += red[1][k]; g11 += red[2][k]; }
        g00 += 1e-6f; g11 += 1e-6f;                    // ridge
        float a0 = g11 - g01, a1 = g00 - g01;          // G^{-1} 1, det cancels in norm
        float ss = a0 + a1;
        if (fabsf(ss) < 1e-20f) { a0 = 0.5f; a1 = 0.5f; }
        else { a0 /= ss; a1 /= ss; }
        alphas[0] = a0; alphas[1] = a1;
    }
    __syncthreads();
    float a0 = alphas[0], a1 = alphas[1];
    for (int rr = 0; rr < 4; ++rr) {
        int gi = hd * NN + threadIdx.x + rr * 256;
        t[gi] = 0.9f * (a0 * X0[gi] + a1 * X1[gi]) + 0.1f * (a0 * F0[gi] + a1 * F1[gi]);
    }
}

// mags = 0.25 * (V_inv + V_inv^T) h = 0.25 * it * (2h + J@g + J^T@g) at t*
__global__ void mags_kernel(const float* __restrict__ xnf, const float* __restrict__ t,
                            const float* __restrict__ Aout, const float* __restrict__ Bout,
                            float* __restrict__ out) {
    int idx = blockIdx.x * 256 + threadIdx.x;
    int d  = idx & 63;
    int i  = (idx >> 6) & 1023;
    int hd = idx >> 16;
    float it = 1.f / t[hd * NN + i];
    float h  = xnf[(size_t)i * DIM + hd * DH + d];
    float a  = Aout[((size_t)hd << 16) + (size_t)i * DH + d];
    float b  = Bout[((size_t)hd << 16) + (size_t)i * DH + d];
    out[(size_t)i * DIM + hd * DH + d] = 0.25f * it * (2.f * h + a + b);
}

// ---------------- host launcher ----------------
extern "C" void kernel_launch(void* const* d_in, const int* in_sizes, int n_in,
                              void* d_out, int out_size, void* d_ws, size_t ws_size,
                              hipStream_t stream) {
    const float* x    = (const float*)d_in[0];
    const float* ln_w = (const float*)d_in[1];
    const float* ln_b = (const float*)d_in[2];
    const float* Wq   = (const float*)d_in[3];
    const float* Wk   = (const float*)d_in[4];
    // d_in[5] causal_mask: implicit (lower triangular)
    float* out = (float*)d_out;

    char* ws = (char*)d_ws;
    size_t off = 0;
    auto alloc = [&](size_t bytes) -> void* {
        off = (off + 255) & ~(size_t)255;
        void* p = ws + off;
        off += bytes;
        return p;
    };
    float*          xnf  = (float*)alloc((size_t)NN * DIM * 4);
    unsigned short* xnb  = (unsigned short*)alloc((size_t)NN * DIM * 2);
    unsigned short* WqT  = (unsigned short*)alloc((size_t)DIM * DIM * 2);
    unsigned short* WkT  = (unsigned short*)alloc((size_t)DIM * DIM * 2);
    unsigned short* qb   = (unsigned short*)alloc((size_t)NN * DIM * 2);
    unsigned short* kb2  = (unsigned short*)alloc((size_t)NN * DIM * 2);
    unsigned short* Jb   = (unsigned short*)alloc((size_t)HEADS * NN * NN * 2);
    unsigned short* JTb  = (unsigned short*)alloc((size_t)HEADS * NN * NN * 2);
    float*          Jd   = (float*)alloc((size_t)HEADS * NN * 4);
    float*          tarr = (float*)alloc((size_t)HEADS * NN * 4);
    float*          X0   = (float*)alloc((size_t)HEADS * NN * 4);
    float*          X1   = (float*)alloc((size_t)HEADS * NN * 4);
    float*          F0   = (float*)alloc((size_t)HEADS * NN * 4);
    float*          F1   = (float*)alloc((size_t)HEADS * NN * 4);
    unsigned short* gT   = (unsigned short*)alloc((size_t)HEADS * DH * NN * 2);
    float*          Aout = (float*)alloc((size_t)HEADS * NN * DH * 4);
    float*          Bout = (float*)alloc((size_t)HEADS * NN * DH * 4);
    (void)ws_size; (void)in_sizes; (void)n_in; (void)out_size;

    ln_kernel<<<NN, 256, 0, stream>>>(x, ln_w, ln_b, xnf, xnb);
    transw_kernel<<<2 * DIM * DIM / 256, 256, 0, stream>>>(Wq, Wk, WqT, WkT);
    qk_gemm_kernel<<<512, 256, 0, stream>>>(xnb, WqT, WkT, qb, kb2);
    attn_kernel<<<HEADS * 64, 256, 0, stream>>>(qb, kb2, Jb, JTb, Jd);
    init_t_kernel<<<(HEADS * NN + 255) / 256, 256, 0, stream>>>(tarr);

    // init history: f0 = f(t0)
    prep_g_kernel<<<HEADS * NN * DH / 256, 256, 0, stream>>>(xnf, tarr, gT);
    matvec_kernel<<<64, 256, 0, stream>>>(Jb, JTb, gT, Aout, Bout);
    solve_kernel<<<HEADS, 256, 0, stream>>>(xnf, Jd, Aout, Bout, tarr, X0, X1, F0, F1, 0);

    for (int it = 0; it < 20; ++it) {
        prep_g_kernel<<<HEADS * NN * DH / 256, 256, 0, stream>>>(xnf, tarr, gT);
        matvec_kernel<<<64, 256, 0, stream>>>(Jb, JTb, gT, Aout, Bout);
        solve_kernel<<<HEADS, 256, 0, stream>>>(xnf, Jd, Aout, Bout, tarr, X0, X1, F0, F1, 1);
    }

    // final sweep at t*, then magnetizations
    prep_g_kernel<<<HEADS * NN * DH / 256, 256, 0, stream>>>(xnf, tarr, gT);
    matvec_kernel<<<64, 256, 0, stream>>>(Jb, JTb, gT, Aout, Bout);
    mags_kernel<<<HEADS * NN * DH / 256, 256, 0, stream>>>(xnf, tarr, Aout, Bout, out);
}
